// StabilizedGATLayer_10514079941116
// MI455X (gfx1250) — matrix operand order repaired and checked by reading, not compiled, not run on hardware
//
#include <hip/hip_runtime.h>

typedef __attribute__((ext_vector_type(16))) __bf16        v16bf;
typedef __attribute__((ext_vector_type(2)))  __bf16        v2bf;
typedef __attribute__((ext_vector_type(2)))  float         v2f;
typedef __attribute__((ext_vector_type(8)))  float         v8f;
typedef __attribute__((ext_vector_type(8)))  unsigned int  v8u;

#define NN 2048
#define HH 4
#define HD 32
#define FF 128
#define WPB 4   // waves per block in the attention kernel

// Native bf16 conversions -> v_cvt_pk_bf16_f32-class ops.
__device__ __forceinline__ unsigned short f2bf(float f) {
    __bf16 b = (__bf16)f;
    return __builtin_bit_cast(unsigned short, b);
}
__device__ __forceinline__ unsigned int pack2bf(float lo, float hi) {
    v2f t; t[0] = lo; t[1] = hi;
    return __builtin_bit_cast(unsigned int, __builtin_convertvector(t, v2bf));
}
__device__ __forceinline__ float bf2f(unsigned short u) {
    return (float)__builtin_bit_cast(__bf16, u);
}
// A-operand K index for VGPR v, lane-half g (16-bit A 16x32 layout)
__device__ __forceinline__ int kA(int v, int g) {
    return (v < 4) ? (2 * v + 8 * g) : (16 + 2 * (v - 4) + 8 * g);
}

// ---------------- Kernel 1: h = x @ W^T  (bf16 out) ----------------
__global__ __launch_bounds__(32) void gemm_h_kernel(
    const float* __restrict__ x, const float* __restrict__ W,
    unsigned short* __restrict__ hbf) {
    const int L = threadIdx.x;
    const int m = L & 15, g = L >> 4;
    const int i0 = blockIdx.x * 16;   // row tile in flat (B*N)
    const int n0 = blockIdx.y * 16;   // output-feature tile
    const float* xrow = x + (long)(i0 + m) * FF;
    const float* wrow = W + (long)(n0 + m) * FF;   // B[k][n] = W[n0+n][k], n==m
    v8f acc = {};
#pragma unroll
    for (int kc = 0; kc < 4; ++kc) {
        const int k0 = kc * 32;
        v8u ap, bp;
#pragma unroll
        for (int v = 0; v < 8; ++v) {
            int ka = kA(v, g);
            float2 xa = *(const float2*)(xrow + k0 + ka);
            ap[v] = pack2bf(xa.x, xa.y);
            int kb = kA(v, g);
            float2 wb = *(const float2*)(wrow + k0 + kb);
            bp[v] = pack2bf(wb.x, wb.y);
        }
        acc = __builtin_amdgcn_wmma_f32_16x16x32_bf16(
            false, __builtin_bit_cast(v16bf, ap),
            false, __builtin_bit_cast(v16bf, bp),
            (short)0, acc, false, false);
    }
#pragma unroll
    for (int r = 0; r < 8; ++r) {
        int row = i0 + r + 8 * g;                 // C layout: M = r + 8g, N = m
        hbf[(long)row * FF + n0 + m] = f2bf(acc[r]);
    }
}

// ---------------- Kernel 2: e_i, e_j per (b, i, head) ----------------
__global__ void eiej_kernel(const unsigned short* __restrict__ hbf,
                            const float* __restrict__ a,
                            float* __restrict__ ei, float* __restrict__ ej) {
    int idx = blockIdx.x * blockDim.x + threadIdx.x;   // over BN*H
    int head = idx & (HH - 1);
    int row  = idx >> 2;                               // flat b*N + i
    int b = row / NN, i = row % NN;
    const unsigned short* hr = hbf + (long)row * FF + head * HD;
    float sl = 0.f, sr = 0.f;
#pragma unroll 8
    for (int d = 0; d < HD; ++d) {
        float hv = bf2f(hr[d]);
        sl += hv * a[d];
        sr += hv * a[HD + d];
    }
    long o = ((long)b * HH + head) * NN + i;
    ei[o] = sl;
    ej[o] = sr;
}

// ---------------- Kernel 3: fused mask+LeakyReLU+exp + P@h via WMMA ----------
// 4 waves / block; each wave owns one 16-row i-tile, all share (b,h) and one
// double-buffered h-tile in LDS. Row softmax denominators come from a third
// WMMA against an all-ones B matrix (identical C-layout slots -> pure
// elementwise normalize, zero cross-lane reductions).
__global__ __launch_bounds__(WPB * 32) void gat_attn_kernel(
    const int* __restrict__ adj, const unsigned short* __restrict__ hbf,
    const float* __restrict__ ei, const float* __restrict__ ej,
    float* __restrict__ out) {
    const int tid = threadIdx.x;
    const int L = tid & 31;
    const int w = tid >> 5;
    const int blk = blockIdx.x;                 // B*H*(N/16)/WPB = 512 blocks
    const int ig = blk & (NN / 16 / WPB - 1);   // 0..31
    const int h  = (blk >> 5) & (HH - 1);
    const int b  = blk >> 7;
    const int i0 = (ig * WPB + w) * 16;

    __shared__ float s_ei[WPB][16];
    __shared__ unsigned short sP[2][WPB][16 * 34];  // P[m][k], stride 34
    __shared__ unsigned short sHT[2][32 * 34];      // HT[d][k] = h[j0+k][d]

    const long ebase = ((long)b * HH + h) * NN;
    if (L < 16) s_ei[w][L] = ei[ebase + i0 + L];
    __syncthreads();
    float eir[16];
#pragma unroll
    for (int r = 0; r < 16; ++r) eir[r] = s_ei[w][r];

    const int m = L & 15, g = L >> 4;
    const int d_s  = tid & 31;          // sHT staging: this thread's d column
    const int r0_s = (tid >> 5) * 8;    // and its 8 j-rows
    v8f acc0 = {}, acc1 = {}, accS = {};
    v8u onesu;
#pragma unroll
    for (int v = 0; v < 8; ++v) onesu[v] = 0x3F803F80u;   // bf16 1.0 pairs
    const v16bf ones = __builtin_bit_cast(v16bf, onesu);

    const long adjb = (long)b * NN * NN;
    for (int jc = 0; jc < 64; ++jc) {
        const int j0 = jc * 32;
        const int p = jc & 1;
        const float ejv = ej[ebase + j0 + L];
        // P tile (per wave): coalesced 128B adj reads, fused lrelu+mask+exp
#pragma unroll 4
        for (int r = 0; r < 16; ++r) {
            int av = adj[adjb + (long)(i0 + r) * NN + j0 + L];
            float e = eir[r] + ejv;
            e = e > 0.f ? e : 0.2f * e;
            float pr = av ? __expf(e) : 0.f;
            sP[p][w][r * 34 + L] = f2bf(pr);
        }
        // shared transposed h tile: 8 rows per thread (4x amortized per wave)
#pragma unroll
        for (int r = 0; r < 8; ++r) {
            sHT[p][d_s * 34 + r0_s + r] =
                hbf[((long)(b * NN + j0 + r0_s + r)) * FF + h * HD + d_s];
        }
        // one instruction prefetches the whole next adj tile (per-lane lines)
        {
            int jn = (jc < 63) ? j0 + 32 : 0;
            __builtin_prefetch(adj + adjb + (long)(i0 + m) * NN + jn + g * 16, 0, 1);
        }
        __syncthreads();   // single barrier: double buffers handle WAR
        v8u ap, b0, b1;
#pragma unroll
        for (int v = 0; v < 8; ++v) {
            ap[v] = *(const unsigned int*)&sP[p][w][m * 34 + kA(v, g)];
            int kb = kA(v, g);                               // B layout K=kA(v,g)
            b0[v] = *(const unsigned int*)&sHT[p][m * 34 + kb];          // d=n
            b1[v] = *(const unsigned int*)&sHT[p][(m + 16) * 34 + kb];   // d=n+16
        }
        const v16bf A = __builtin_bit_cast(v16bf, ap);
        acc0 = __builtin_amdgcn_wmma_f32_16x16x32_bf16(
            false, A, false, __builtin_bit_cast(v16bf, b0), (short)0, acc0, false, false);
        acc1 = __builtin_amdgcn_wmma_f32_16x16x32_bf16(
            false, A, false, __builtin_bit_cast(v16bf, b1), (short)0, acc1, false, false);
        accS = __builtin_amdgcn_wmma_f32_16x16x32_bf16(
            false, A, false, ones, (short)0, accS, false, false);
    }
    // normalize: accS holds row sums in identical C-layout slots
#pragma unroll
    for (int r = 0; r < 8; ++r) {
        const int row = i0 + r + 8 * g;
        const float inv = 1.0f / accS[r];
        const long o = ((long)(b * NN + row)) * FF + h * HD + m;
        out[o]      = acc0[r] * inv;
        out[o + 16] = acc1[r] * inv;
    }
}

extern "C" void kernel_launch(void* const* d_in, const int* in_sizes, int n_in,
                              void* d_out, int out_size, void* d_ws, size_t ws_size,
                              hipStream_t stream) {
    const float* x   = (const float*)d_in[0];
    const int*   adj = (const int*)d_in[1];
    const float* W   = (const float*)d_in[2];
    const float* a   = (const float*)d_in[3];

    const long BN = 4L * NN;                       // 8192 rows
    unsigned short* hbf = (unsigned short*)d_ws;   // BN*128 bf16 = 2 MB
    float* ei = (float*)((char*)d_ws + BN * FF * sizeof(unsigned short));
    float* ej = ei + 4L * HH * NN;

    gemm_h_kernel<<<dim3((int)(BN / 16), FF / 16), 32, 0, stream>>>(x, W, hbf);
    eiej_kernel<<<(int)(BN * HH / 256), 256, 0, stream>>>(hbf, a, ei, ej);
    gat_attn_kernel<<<(int)(4L * HH * (NN / 16) / WPB), WPB * 32, 0, stream>>>(
        adj, hbf, ei, ej, (float*)d_out);
}
